// MMDLoss_78932908965970
// MI455X (gfx1250) — compile-verified, matrix operand order checked by reading, not run on hardware
//
#include <hip/hip_runtime.h>

// MMD RBF loss, z: [8192, 256] f32, SIGMA=1 -> gamma=0.5.
// rowsq -> bf16-WMMA tiled pairwise exp-sum (upper triangle x2, branch-free
// epilogue) -> finalize (partials reduce + separable diagonal correction +
// analytic terms).

typedef __attribute__((ext_vector_type(16))) __bf16       v16bf;
typedef __attribute__((ext_vector_type(8)))  float        v8f;
typedef __attribute__((ext_vector_type(4)))  float        f32x4;
typedef __attribute__((ext_vector_type(4)))  unsigned int u32x4;

#define N_PTS 8192
#define DIM   256
#define BT    128            // block tile edge (rows and cols)
#define KT    64             // K chunk per LDS stage (two bf16 WMMA K-steps)
#define LDT   72             // LDS tile row stride in bf16 elems (64 + 8 pad)
#define NB    (N_PTS / BT)   // 64 block tiles per side

#define WMMA_BF16(a, b, c) \
  __builtin_amdgcn_wmma_f32_16x16x32_bf16(false, (a), false, (b), (short)0, (c), false, false)

// round-to-nearest-even f32 -> bf16, packed pair
__device__ __forceinline__ unsigned int pack_bf16x2(float a, float b) {
  unsigned int ua = __float_as_uint(a);
  unsigned int ub = __float_as_uint(b);
  ua = (ua + 0x7FFFu + ((ua >> 16) & 1u)) >> 16;
  ub = (ub + 0x7FFFu + ((ub >> 16) & 1u)) & 0xFFFF0000u;
  return ua | ub;
}

// f32 value after round-trip through bf16 (for the diagonal correction)
__device__ __forceinline__ float bf16_round(float a) {
  unsigned int u = __float_as_uint(a);
  u = (u + 0x7FFFu + ((u >> 16) & 1u)) & 0xFFFF0000u;
  return __uint_as_float(u);
}

// Load one 128-row x 64-K f32 tile from global, convert to bf16, store to LDS.
// 256 threads: thread t handles row t/2, 32-float half (t&1).
__device__ __forceinline__ void load_tile(unsigned short* dst,
                                          const float* __restrict__ z,
                                          int rowBase, int k0, int tid) {
  int r  = tid >> 1;
  int ch = (tid & 1) << 5;  // 0 or 32 (floats / bf16 elems)
  const f32x4* src = (const f32x4*)(z + (size_t)(rowBase + r) * DIM + k0 + ch);
  u32x4* d = (u32x4*)&dst[r * LDT + ch];  // byte addr r*144 + ch*2, 16B aligned
#pragma unroll
  for (int q = 0; q < 4; ++q) {
    f32x4 f0 = src[2 * q];
    f32x4 f1 = src[2 * q + 1];
    u32x4 p;
    p.x = pack_bf16x2(f0.x, f0.y);
    p.y = pack_bf16x2(f0.z, f0.w);
    p.z = pack_bf16x2(f1.x, f1.y);
    p.w = pack_bf16x2(f1.z, f1.w);
    d[q] = p;
  }
}

// Load a 16x32 bf16 operand fragment (WMMA A layout; same layout serves B of
// z*z^T since B[k][n] = z[colBase+n][k]).
// lane<16: row=r0+lane, K {0..7} in v0-3, {16..23} in v4-7
// lane>=16: row=r0+lane-16, K {8..15} and {24..31}
__device__ __forceinline__ v16bf load_frag(const unsigned short* tile, int r0,
                                           int ks, int lid) {
  int row = r0 + (lid & 15);
  int kb  = ks + ((lid >> 4) << 3);  // ks + (0 or 8)
  union { v16bf v; u32x4 q[2]; } f;
  f.q[0] = *(const u32x4*)&tile[row * LDT + kb];
  f.q[1] = *(const u32x4*)&tile[row * LDT + kb + 16];
  return f.v;
}

// ---------------- kernel 1: per-row squared norms ----------------
__global__ __launch_bounds__(256) void rowsq_kernel(const float* __restrict__ z,
                                                    float* __restrict__ sq) {
  int row  = (blockIdx.x * blockDim.x + threadIdx.x) >> 5;  // one wave32 per row
  int lane = threadIdx.x & 31;
  const f32x4* zr = (const f32x4*)(z + (size_t)row * DIM);
  f32x4 a = zr[lane], b = zr[lane + 32];
  float s = a.x * a.x + a.y * a.y + a.z * a.z + a.w * a.w +
            b.x * b.x + b.y * b.y + b.z * b.z + b.w * b.w;
#pragma unroll
  for (int o = 16; o > 0; o >>= 1) s += __shfl_xor(s, o, 32);
  if (lane == 0) sq[row] = s;
}

// ---------------- kernel 2: tiled pairwise exp(-gamma*dist^2) sum ----------------
__global__ __launch_bounds__(256) void mmd_pair_kernel(const float* __restrict__ z,
                                                       const float* __restrict__ sq,
                                                       float* __restrict__ partials) {
  int bj = blockIdx.x, bi = blockIdx.y;
  int tid = threadIdx.x;
  // symmetry: only bj >= bi, off-diagonal weighted x2; skipped blocks write 0
  if (bj < bi) {
    if (tid == 0) partials[bi * NB + bj] = 0.0f;
    return;
  }

  __shared__ __align__(16) unsigned short lA[BT * LDT];
  __shared__ __align__(16) unsigned short lB[BT * LDT];
  __shared__ __align__(16) float sSqR[BT];
  __shared__ __align__(16) float sSqC[BT];
  __shared__ float red[256];

  int rowBase = bi * BT, colBase = bj * BT;
  if (tid < BT) sSqR[tid] = sq[rowBase + tid];
  else          sSqC[tid - BT] = sq[colBase + tid - BT];

  int w   = tid >> 5;        // wave 0..7
  int lid = tid & 31;
  int wr  = (w >> 1) * 32;   // wave row offset: 0/32/64/96
  int wc  = (w & 1) * 64;    // wave col offset: 0/64

  v8f acc[2][4] = {};

  for (int kt = 0; kt < DIM / KT; ++kt) {
    int k0 = kt * KT;
    load_tile(lA, z, rowBase, k0, tid);
    load_tile(lB, z, colBase, k0, tid);
    __syncthreads();
#pragma unroll
    for (int ks = 0; ks < KT; ks += 32) {
      // load all six fragments first so WMMAs can overlap the LDS loads
      v16bf a0 = load_frag(lA, wr,      ks, lid);
      v16bf a1 = load_frag(lA, wr + 16, ks, lid);
      v16bf b0 = load_frag(lB, wc,      ks, lid);
      v16bf b1 = load_frag(lB, wc + 16, ks, lid);
      v16bf b2 = load_frag(lB, wc + 32, ks, lid);
      v16bf b3 = load_frag(lB, wc + 48, ks, lid);
      acc[0][0] = WMMA_BF16(a0, b0, acc[0][0]);
      acc[0][1] = WMMA_BF16(a0, b1, acc[0][1]);
      acc[0][2] = WMMA_BF16(a0, b2, acc[0][2]);
      acc[0][3] = WMMA_BF16(a0, b3, acc[0][3]);
      acc[1][0] = WMMA_BF16(a1, b0, acc[1][0]);
      acc[1][1] = WMMA_BF16(a1, b1, acc[1][1]);
      acc[1][2] = WMMA_BF16(a1, b2, acc[1][2]);
      acc[1][3] = WMMA_BF16(a1, b3, acc[1][3]);
    }
    __syncthreads();
  }

  // Branch-free epilogue: dist^2 = |zi|^2 + |zj|^2 - 2*gram, clamp >= 0,
  // accumulate exp(-0.5*dist^2). Diagonal handled in finalize_kernel.
  // C/D layout: VGPR e, lanes 0-15 -> M=e, lanes 16-31 -> M=e+8; N = lane&15
  int nLoc = lid & 15;
  int mHi  = (lid >> 4) << 3;
  float lsum = 0.0f;
#pragma unroll
  for (int rb = 0; rb < 2; ++rb) {
    int rl = wr + rb * 16 + mHi;              // multiple of 8 -> 32B aligned
    f32x4 s0 = *(const f32x4*)&sSqR[rl];
    f32x4 s1 = *(const f32x4*)&sSqR[rl + 4];
    float sqi[8] = {s0.x, s0.y, s0.z, s0.w, s1.x, s1.y, s1.z, s1.w};
#pragma unroll
    for (int cb = 0; cb < 4; ++cb) {
      float sqj = sSqC[wc + cb * 16 + nLoc];
#pragma unroll
      for (int e = 0; e < 8; ++e) {
        float d = fmaxf(sqi[e] + sqj - 2.0f * acc[rb][cb][e], 0.0f);
        lsum += __expf(-0.5f * d);            // gamma = 0.5
      }
    }
  }

  red[tid] = lsum;
  __syncthreads();
  for (int s = 128; s > 0; s >>= 1) {
    if (tid < s) red[tid] += red[tid + s];
    __syncthreads();
  }
  if (tid == 0) partials[bi * NB + bj] = red[0] * ((bj > bi) ? 2.0f : 1.0f);
}

// ---------------- kernel 3: reduce + diagonal fix + analytic terms ----------------
__global__ __launch_bounds__(256) void finalize_kernel(const float* __restrict__ z,
                                                       const float* __restrict__ partials,
                                                       const float* __restrict__ sq,
                                                       float* __restrict__ out) {
  __shared__ float red[256];
  int t = threadIdx.x;

  float sA = 0.0f;
  for (int i = t; i < NB * NB; i += 256) sA += partials[i];

  // Diagonal correction: replace exp(-0.5*d_ii_bf16) with exact exp(0)=1.
  // Recompute d_ii from bf16-rounded z (matches WMMA inputs; accumulation-order
  // differences are ~1e-7 and irrelevant).
  float sC = 0.0f;
  for (int i = t; i < N_PTS; i += 256) {
    const f32x4* zi = (const f32x4*)(z + (size_t)i * DIM);
    float g = 0.0f;
#pragma unroll 4
    for (int k = 0; k < DIM / 4; ++k) {
      f32x4 v = zi[k];
      float x = bf16_round(v.x), y = bf16_round(v.y);
      float zz = bf16_round(v.z), ww = bf16_round(v.w);
      g += x * x + y * y + zz * zz + ww * ww;
    }
    float d = fmaxf(2.0f * sq[i] - 2.0f * g, 0.0f);
    sC += 1.0f - __expf(-0.5f * d);
  }

  float sB = 0.0f;
  for (int i = t; i < N_PTS; i += 256) sB += __expf(-0.25f * sq[i]);  // gamma/denom_b

  red[t] = sA + sC;
  __syncthreads();
  for (int s = 128; s > 0; s >>= 1) { if (t < s) red[t] += red[t + s]; __syncthreads(); }
  float sumA = red[0];
  __syncthreads();
  red[t] = sB;
  __syncthreads();
  for (int s = 128; s > 0; s >>= 1) { if (t < s) red[t] += red[t + s]; __syncthreads(); }

  if (t == 0) {
    double termA = (double)sumA / ((double)N_PTS * (double)N_PTS);
    double coeffB = 2.938735877055719e-39;  // (1+2*gamma)^(-d/2) = 2^-128
    double termB = coeffB * ((double)red[0] / (double)N_PTS);
    double termC = 8.481654e-62;            // (1+4*gamma)^(-d/2) = 3^-128
    out[0] = (float)(termA - 2.0 * termB + termC);
  }
}

extern "C" void kernel_launch(void* const* d_in, const int* in_sizes, int n_in,
                              void* d_out, int out_size, void* d_ws, size_t ws_size,
                              hipStream_t stream) {
  const float* z = (const float*)d_in[0];
  float* out = (float*)d_out;
  float* partials = (float*)d_ws;                                   // NB*NB floats (16 KB)
  float* sq = (float*)((char*)d_ws + NB * NB * sizeof(float));      // N_PTS floats (32 KB)

  rowsq_kernel<<<N_PTS / 8, 256, 0, stream>>>(z, sq);
  dim3 grid(NB, NB);
  mmd_pair_kernel<<<grid, 256, 0, stream>>>(z, sq, partials);
  finalize_kernel<<<1, 256, 0, stream>>>(z, partials, sq, out);
}
